// CrossAttentionAVG_58677843198645
// MI455X (gfx1250) — matrix operand order repaired and checked by reading, not compiled, not run on hardware
//
#include <hip/hip_runtime.h>
#include <hip/hip_bf16.h>

typedef __bf16 bf16;
typedef __attribute__((ext_vector_type(16))) __bf16 v16bf;
typedef __attribute__((ext_vector_type(8)))  __bf16 v8bf;
typedef __attribute__((ext_vector_type(4)))  __bf16 v4bf;
typedef __attribute__((ext_vector_type(8)))  float  v8f;

union FragBF { v16bf v; v8bf h[2]; };

#define BM 128
#define BN 128
#define BK 32
#define LDSP 40   // padded K-stride (bf16 elems): 80B rows -> conflict-free b128 reads

// CDNA5 async DMA: memory -> LDS with no VGPR round trip, tracked by ASYNCcnt.
// %0 = 32-bit LDS byte address (low 32 bits of generic shared pointer, per ISA
// aperture rule); %1 = 64-bit global address (VGPR pair).
__device__ __forceinline__ void async_b128_to_lds(const bf16* g, bf16* l) {
    asm volatile("global_load_async_to_lds_b128 %0, %1, off"
                 :: "v"((unsigned)(unsigned long long)l),
                    "v"((unsigned long long)g)
                 : "memory");
}

__device__ __forceinline__ void wait_asynccnt0() {
    asm volatile("s_wait_asynccnt 0" ::: "memory");
}

// ---------------------------------------------------------------------------
// f32 -> bf16 cast (vectorized x4, grid-stride)
// ---------------------------------------------------------------------------
__global__ void __launch_bounds__(256)
cast_f32_to_bf16(const float* __restrict__ in, bf16* __restrict__ out, size_t n4) {
    size_t i = (size_t)blockIdx.x * blockDim.x + threadIdx.x;
    const size_t stride = (size_t)gridDim.x * blockDim.x;
    for (; i < n4; i += stride) {
        const float4 f = ((const float4*)in)[i];
        v4bf o;
        o.x = (bf16)f.x; o.y = (bf16)f.y; o.z = (bf16)f.z; o.w = (bf16)f.w;
        ((v4bf*)out)[i] = o;
    }
}

// ---------------------------------------------------------------------------
// Batched NT GEMM: D[b] = alpha * A[b] (MxK, row-major) * B[b]^T (B is NxK row-major)
// Requires: M%128==0, N%128==0, K%32==0, blockDim=256 (8 waves, wave32).
// Wave layout: 4 waves along M (32 rows each), 2 along N (64 cols each).
// Each wave: 2x4 grid of 16x16 WMMA tiles -> 8 v8f accumulators.
// Global->LDS staging uses global_load_async_to_lds_b128 (ASYNCcnt), double
// buffered so the DMA of tile k+1 overlaps the 8 WMMAs of tile k.
// Output: f32 to Df if Dh==nullptr, else bf16 to Dh.
// ---------------------------------------------------------------------------
__global__ void __launch_bounds__(256)
gemm_nt_bf16(const bf16* __restrict__ A, const bf16* __restrict__ Bm,
             float* __restrict__ Df, bf16* __restrict__ Dh,
             int K, int lda, int ldb, int ldd,
             long long sA, long long sB, long long sD, float alpha)
{
    __shared__ bf16 As[2][BM * LDSP];
    __shared__ bf16 Bs[2][BM * LDSP];

    const int tid  = threadIdx.x;
    const int lane = tid & 31;
    const int wave = tid >> 5;
    const int bz   = blockIdx.z;

    const bf16* Ab = A  + (size_t)bz * sA;
    const bf16* Bb = Bm + (size_t)bz * sB;
    const int tileM = blockIdx.y * BM;
    const int tileN = blockIdx.x * BN;

    // global->LDS staging: each thread issues 2x(A,B) async b128 copies per tile
    const int lr = tid >> 2;          // row 0..63 (two passes cover 128 rows)
    const int lc = (tid & 3) << 3;    // bf16 col 0/8/16/24 within 32-wide K strip

    auto issue_tile = [&](int buf, int k0) {
#pragma unroll
        for (int p = 0; p < 2; ++p) {
            const int row = lr + (p << 6);
            async_b128_to_lds(Ab + (size_t)(tileM + row) * lda + (k0 + lc),
                              &As[buf][row * LDSP + lc]);
            async_b128_to_lds(Bb + (size_t)(tileN + row) * ldb + (k0 + lc),
                              &Bs[buf][row * LDSP + lc]);
        }
    };

    v8f acc[2][4];
#pragma unroll
    for (int mt = 0; mt < 2; ++mt)
#pragma unroll
        for (int nt = 0; nt < 4; ++nt)
            acc[mt][nt] = (v8f){};

    const int wm   = (wave & 3) << 5;   // wave M offset: 0/32/64/96
    const int wn   = (wave >> 2) << 6;  // wave N offset: 0/64
    const int frow = lane & 15;         // matrix row (A: M, B: N) within 16x16 tile
    const int akb  = (lane >> 4) << 3;  // A frag K base: 0 or 8   (ISA 7.12.2 16-bit A)
    const int bkb  = (lane >> 4) << 3;  // B frag K base: 0 or 8   (ISA 7.12.2 B layout)

    issue_tile(0, 0);
    wait_asynccnt0();
    __syncthreads();

    const int KT = K / BK;
    for (int kt = 0; kt < KT; ++kt) {
        const int cur = kt & 1;
        // Kick off DMA of the next tile; it runs while we do the WMMAs below.
        if (kt + 1 < KT) issue_tile(cur ^ 1, (kt + 1) * BK);
        if (kt + 2 < KT) {  // global_prefetch_b8 on K+2 tile into GL2
            __builtin_prefetch(Ab + (size_t)(tileM + lr) * lda + (kt + 2) * BK + lc, 0, 1);
            __builtin_prefetch(Bb + (size_t)(tileN + lr) * ldb + (kt + 2) * BK + lc, 0, 1);
        }

        FragBF a[2], b[4];
#pragma unroll
        for (int mt = 0; mt < 2; ++mt) {
            const bf16* pa = &As[cur][(wm + mt * 16 + frow) * LDSP + akb];
            a[mt].h[0] = *(const v8bf*)pa;          // K = kb+0..7
            a[mt].h[1] = *(const v8bf*)(pa + 16);   // K = kb+16..23
        }
#pragma unroll
        for (int nt = 0; nt < 4; ++nt) {
            const bf16* pb = &Bs[cur][(wn + nt * 16 + frow) * LDSP + bkb];
            b[nt].h[0] = *(const v8bf*)pb;          // K = kb+0..7
            b[nt].h[1] = *(const v8bf*)(pb + 16);   // K = kb+16..23
        }
#pragma unroll
        for (int mt = 0; mt < 2; ++mt)
#pragma unroll
            for (int nt = 0; nt < 4; ++nt)
                acc[mt][nt] = __builtin_amdgcn_wmma_f32_16x16x32_bf16(
                    false, a[mt].v, false, b[nt].v, (short)0, acc[mt][nt], false, false);

        // Drain this wave's async DMAs, then barrier so everyone sees the tile.
        wait_asynccnt0();
        __syncthreads();
    }

    // Epilogue: C/D layout -> lane n = lane&15, m = (lane>>4)*8 + vgpr_j
    const long long dbase = (long long)bz * sD;
#pragma unroll
    for (int mt = 0; mt < 2; ++mt) {
#pragma unroll
        for (int nt = 0; nt < 4; ++nt) {
            const int n  = tileN + wn + nt * 16 + (lane & 15);
            const int mb = tileM + wm + mt * 16 + ((lane >> 4) << 3);
#pragma unroll
            for (int j = 0; j < 8; ++j) {
                const float v = acc[mt][nt][j] * alpha;
                const size_t idx = (size_t)(dbase + (long long)(mb + j) * ldd + n);
                if (Dh) Dh[idx] = (bf16)v;
                else    Df[idx] = v;
            }
        }
    }
}

// ---------------------------------------------------------------------------
// Row softmax over S=2048: one block per row, 8 elems/thread.
// In-place f32 normalize + bf16 copy for the context GEMM.
// ---------------------------------------------------------------------------
__global__ void __launch_bounds__(256)
softmax_rows(float* __restrict__ attn, bf16* __restrict__ attn_bf) {
    const int row  = blockIdx.x;
    const int tid  = threadIdx.x;
    const int lane = tid & 31;
    const int wave = tid >> 5;
    float* p = attn + (size_t)row * 2048;

    float x[8];
    float m = -3.0e38f;
#pragma unroll
    for (int i = 0; i < 8; ++i) { x[i] = p[tid + (i << 8)]; m = fmaxf(m, x[i]); }
#pragma unroll
    for (int off = 16; off > 0; off >>= 1) m = fmaxf(m, __shfl_xor(m, off, 32));

    __shared__ float redm[8];
    __shared__ float reds[8];
    if (lane == 0) redm[wave] = m;
    __syncthreads();
    m = redm[0];
#pragma unroll
    for (int w = 1; w < 8; ++w) m = fmaxf(m, redm[w]);

    float s = 0.f;
#pragma unroll
    for (int i = 0; i < 8; ++i) { x[i] = __expf(x[i] - m); s += x[i]; }
#pragma unroll
    for (int off = 16; off > 0; off >>= 1) s += __shfl_xor(s, off, 32);
    if (lane == 0) reds[wave] = s;
    __syncthreads();
    s = 0.f;
#pragma unroll
    for (int w = 0; w < 8; ++w) s += reds[w];

    const float inv = 1.0f / s;
    bf16* pb = attn_bf + (size_t)row * 2048;
#pragma unroll
    for (int i = 0; i < 8; ++i) {
        const float v = x[i] * inv;
        p[tid + (i << 8)]  = v;
        pb[tid + (i << 8)] = (bf16)v;
    }
}

// ---------------------------------------------------------------------------
extern "C" void kernel_launch(void* const* d_in, const int* in_sizes, int n_in,
                              void* d_out, int out_size, void* d_ws, size_t ws_size,
                              hipStream_t stream) {
    (void)in_sizes; (void)n_in; (void)out_size; (void)ws_size;
    constexpr int B = 4, T = 2048, S = 2048, H = 1024;

    const float* tar = (const float*)d_in[0];
    const float* src = (const float*)d_in[1];
    const float* ori = (const float*)d_in[2];
    const float* Wq  = (const float*)d_in[3];
    const float* Wk  = (const float*)d_in[4];
    const float* Wv  = (const float*)d_in[5];

    float* ctx  = (float*)d_out;                       // (B,T,H)
    float* attn = ctx + (size_t)B * T * H;             // (B,T,S)

    const size_t nBTH = (size_t)B * T * H;
    const size_t nBSH = (size_t)B * S * H;
    const size_t nW   = (size_t)H * H;
    const size_t nBTS = (size_t)B * T * S;

    bf16* w = (bf16*)d_ws;
    bf16* tar_h = w; w += nBTH;
    bf16* src_h = w; w += nBSH;
    bf16* ori_h = w; w += nBSH;
    bf16* wq_h  = w; w += nW;
    bf16* wk_h  = w; w += nW;
    bf16* wv_h  = w; w += nW;
    bf16* q_h   = w; w += nBSH;   // (B,S,H)
    bf16* k_h   = w; w += nBTH;   // (B,T,H)
    bf16* vT_h  = w; w += nBSH;   // (B,H,S) = V^T per batch
    bf16* at_h  = w; w += nBTS;   // bf16 attn

    // 1) Down-convert all f32 inputs to bf16 once.
    auto cg = [](size_t n4) { size_t g = (n4 + 255) / 256; if (g > 16384) g = 16384; return dim3((unsigned)g); };
    cast_f32_to_bf16<<<cg(nBTH / 4), 256, 0, stream>>>(tar, tar_h, nBTH / 4);
    cast_f32_to_bf16<<<cg(nBSH / 4), 256, 0, stream>>>(src, src_h, nBSH / 4);
    cast_f32_to_bf16<<<cg(nBSH / 4), 256, 0, stream>>>(ori, ori_h, nBSH / 4);
    cast_f32_to_bf16<<<cg(nW / 4),   256, 0, stream>>>(Wq,  wq_h,  nW / 4);
    cast_f32_to_bf16<<<cg(nW / 4),   256, 0, stream>>>(Wk,  wk_h,  nW / 4);
    cast_f32_to_bf16<<<cg(nW / 4),   256, 0, stream>>>(Wv,  wv_h,  nW / 4);

    // 2) q = src @ Wq^T  (M=B*S, N=H, K=H), bf16 out
    gemm_nt_bf16<<<dim3(H / BN, (B * S) / BM, 1), 256, 0, stream>>>(
        src_h, wq_h, nullptr, q_h, H, H, H, H, 0LL, 0LL, 0LL, 1.0f);
    // 3) k = tar @ Wk^T  (M=B*T, N=H, K=H), bf16 out
    gemm_nt_bf16<<<dim3(H / BN, (B * T) / BM, 1), 256, 0, stream>>>(
        tar_h, wk_h, nullptr, k_h, H, H, H, H, 0LL, 0LL, 0LL, 1.0f);
    // 4) vT[b] = Wv @ ori[b]^T  (M=H, N=S, K=H), bf16 out -> (B,H,S), no transpose kernel needed
    gemm_nt_bf16<<<dim3(S / BN, H / BM, B), 256, 0, stream>>>(
        wv_h, ori_h, nullptr, vT_h, H, H, H, S,
        0LL, (long long)S * H, (long long)H * S, 1.0f);
    // 5) scores[b] = (k[b] @ q[b]^T) / sqrt(H)  (M=T, N=S, K=H), f32 out into d_out attn region
    gemm_nt_bf16<<<dim3(S / BN, T / BM, B), 256, 0, stream>>>(
        k_h, q_h, attn, nullptr, H, H, H, S,
        (long long)T * H, (long long)S * H, (long long)T * S, 1.0f / 32.0f);
    // 6) softmax over s, in place; also emit bf16 attn
    softmax_rows<<<dim3(B * T), 256, 0, stream>>>(attn, at_h);
    // 7) context[b] = attn[b] @ (vT[b])^T  (M=T, N=H, K=S), f32 out
    gemm_nt_bf16<<<dim3(H / BN, T / BM, B), 256, 0, stream>>>(
        at_h, vT_h, ctx, nullptr, S, S, S, H,
        (long long)T * S, (long long)H * S, (long long)T * H, 1.0f);
}
